// Qwen3_5MoeForCausalLM_47382079209674
// MI455X (gfx1250) — matrix operand order repaired and checked by reading, not compiled, not run on hardware
//
#include <hip/hip_runtime.h>
#include <hip/hip_bf16.h>
#include <math.h>

// ---------------- constants from the reference ----------------
constexpr int T_SEQ      = 8192;
constexpr int HIDDEN     = 2048;
constexpr int NUM_V_HEADS= 32;
constexpr int NUM_K_HEADS= 16;
constexpr int HEAD       = 128;              // HEAD_K == HEAD_V
constexpr int KEY_DIM    = NUM_K_HEADS * HEAD;   // 2048
constexpr int VALUE_DIM  = NUM_V_HEADS * HEAD;   // 4096
constexpr int CONV_DIM   = 2 * KEY_DIM + VALUE_DIM; // 8192
constexpr int QKVZ_ROWS  = 2 * KEY_DIM + 2 * VALUE_DIM; // 12288
constexpr float EPS      = 1e-6f;

typedef __attribute__((ext_vector_type(16))) __bf16 v16bf;
typedef __attribute__((ext_vector_type(8)))  __bf16 v8bf;
typedef __attribute__((ext_vector_type(8)))  float  v8f;

// ---------------- fp32 -> bf16 cast (RNE via compiler conversion) -----------
__global__ void k_cast_bf16(const float* __restrict__ src, __bf16* __restrict__ dst, size_t n)
{
    size_t i      = (size_t)blockIdx.x * blockDim.x + threadIdx.x;
    size_t stride = (size_t)gridDim.x * blockDim.x;
    for (; i < n; i += stride) dst[i] = (__bf16)src[i];
}

// ---------------- WMMA bf16 GEMM:  C[M,N] = A[M,K] * B[N,K]^T  --------------
// grid = (N/64, M/128), block = 256 (8 waves). Block tile: 128(M) x 64(N).
// Each wave computes a 16x64 slab, reusing its A fragment over 4 B tiles.
// B tile (64 rows x 32 halves per K-step) is staged cooperatively into LDS
// via GLOBAL_LOAD_ASYNC_TO_LDS_B128, double-buffered on ASYNCcnt: issue the
// next tile, s_wait_asynccnt 1, barrier, compute current from LDS.
// LDS B row pitch = 80B (64B data + 16B pad): fragment reads hit bank
// 20*lane mod 64, distinct per half-wave -> conflict-free ds_load_b128.
// Fragment layouts follow CDNA5 ISA 7.12.2 (wave32):
//   A 16x32 bf16 : lane&15 = row M; halves e=0..7 -> K = e + 8*(lane>>4),
//                  e=8..15 -> K = (e-8) + 8*(lane>>4) + 16
//   B 32x16 bf16 : lane&15 = col N; halves e -> same K as A (e + 8*(lane>>4), +16 for e>=8)
//   C/D 16x16 f32: VGPR r -> M = r + 8*(lane>>4), N = lane&15
constexpr int B_ROW_BYTES  = 80;                 // 64B payload + 16B pad
constexpr int B_TILE_BYTES = 64 * B_ROW_BYTES;   // 5120 B per buffer

__global__ void __launch_bounds__(256) k_gemm_bf16(const __bf16* __restrict__ A,
                                                   const __bf16* __restrict__ B,
                                                   float* __restrict__ C,
                                                   int M, int N, int K)
{
    __shared__ char smemB[2][B_TILE_BYTES];      // 10 KB static LDS

    const int tid  = threadIdx.x;
    const int lane = tid & 31;
    const int wave = tid >> 5;
    const int khi  = lane >> 4;        // 0 or 1 (K half selector)
    const int l15  = lane & 15;
    const int m0   = blockIdx.y * 128 + wave * 16;
    const int n0   = blockIdx.x * 64;

    // async-copy mapping: 256 threads x 16B = 4KB = one 64x32-half B tile
    const int crow = tid >> 2;         // 0..63 : B row within tile
    const int cchk = tid & 3;          // 0..3  : 16B chunk within row
    const __bf16* __restrict__ gB = B + (size_t)(n0 + crow) * K + cchk * 8;
    const unsigned ldsB0   = (unsigned)(size_t)&smemB[0][0];
    const unsigned ldsDoff = (unsigned)(crow * B_ROW_BYTES + cchk * 16);

    const __bf16* __restrict__ arow = A + (size_t)(m0 + l15) * K;

    v8f acc[4] = {v8f{}, v8f{}, v8f{}, v8f{}};

    const int nsteps = K >> 5;         // K / 32

    // prologue: stage first B tile
    {
        unsigned dst = ldsB0 + ldsDoff;
        const void* src = (const void*)gB;
        asm volatile("global_load_async_to_lds_b128 %0, %1, off"
                     :: "v"(dst), "v"(src) : "memory");
    }

    for (int s = 0; s < nsteps; ++s) {
        const int k0 = s << 5;
        if (s + 1 < nsteps) {
            // issue next tile, then wait only for the current one (<=1 pending)
            unsigned dst = ldsB0 + (unsigned)(((s + 1) & 1) * B_TILE_BYTES) + ldsDoff;
            const void* src = (const void*)(gB + k0 + 32);
            asm volatile("global_load_async_to_lds_b128 %0, %1, off"
                         :: "v"(dst), "v"(src) : "memory");
            asm volatile("s_wait_asynccnt 0x1" ::: "memory");
        } else {
            asm volatile("s_wait_asynccnt 0x0" ::: "memory");
        }
        __syncthreads();               // current B tile visible to all waves

        if (k0 + 256 < K) {
            __builtin_prefetch((const void*)(arow + k0 + 256), 0, 1);
        }

        // ---- A fragment from global (row-major, K contiguous) ----
        v16bf a;
        {
            const v8bf a_lo = *(const v8bf*)(arow + k0 + 8 * khi);
            const v8bf a_hi = *(const v8bf*)(arow + k0 + 8 * khi + 16);
#pragma unroll
            for (int e = 0; e < 8; ++e) { a[e] = a_lo[e]; a[e + 8] = a_hi[e]; }
        }

        // ---- 4 B fragments from LDS, 4 WMMAs ----
        const char* bbuf = &smemB[s & 1][0];
#pragma unroll
        for (int nt = 0; nt < 4; ++nt) {
            const char* brow = bbuf + (nt * 16 + l15) * B_ROW_BYTES + khi * 16;
            const v8bf b_lo = *(const v8bf*)(brow);
            const v8bf b_hi = *(const v8bf*)(brow + 32);
            v16bf b;
#pragma unroll
            for (int e = 0; e < 8; ++e) { b[e] = b_lo[e]; b[e + 8] = b_hi[e]; }
            acc[nt] = __builtin_amdgcn_wmma_f32_16x16x32_bf16(
                false, a, false, b, (short)0, acc[nt], false, false);
        }
        __syncthreads();               // all reads done before buffer reuse
    }

    // ---- store D (f32, 16x16 layout) ----
#pragma unroll
    for (int nt = 0; nt < 4; ++nt) {
#pragma unroll
        for (int r = 0; r < 8; ++r) {
            const int row = m0 + r + 8 * khi;
            const int col = n0 + nt * 16 + l15;
            C[(size_t)row * N + col] = acc[nt][r];
        }
    }
}

// ---------------- ba = hs @ W_ba^T, fused into beta / g ---------------------
// grid = T, block = 64 (j<32 -> beta, j>=32 -> g)
__global__ void k_ba_gates(const float* __restrict__ hs,
                           const float* __restrict__ W_ba,
                           const float* __restrict__ dt_bias,
                           const float* __restrict__ A_log,
                           float* __restrict__ beta_out,
                           float* __restrict__ g_out)
{
    const int t = blockIdx.x;
    const int j = threadIdx.x;
    const float4* __restrict__ h = (const float4*)(hs   + (size_t)t * HIDDEN);
    const float4* __restrict__ w = (const float4*)(W_ba + (size_t)j * HIDDEN);
    float acc = 0.f;
#pragma unroll 4
    for (int kk = 0; kk < HIDDEN / 4; ++kk) {
        float4 a = h[kk], b = w[kk];
        acc += a.x * b.x + a.y * b.y + a.z * b.z + a.w * b.w;
    }
    if (j < NUM_V_HEADS) {
        beta_out[(size_t)t * NUM_V_HEADS + j] = 1.f / (1.f + __expf(-acc));
    } else {
        const int hh = j - NUM_V_HEADS;
        const float x  = acc + dt_bias[hh];
        const float sp = (x > 20.f) ? x : log1pf(__expf(x));   // softplus
        g_out[(size_t)t * NUM_V_HEADS + hh] = -__expf(A_log[hh]) * sp;
    }
}

// ------- causal conv(4) + SiLU + per-head l2norm + head repeat --------------
// grid = (T, 64): groups 0..15 q-heads, 16..31 k-heads, 32..63 v-heads.
// block = 128 (one thread per channel in the head).
__global__ void k_prep(const float* __restrict__ mixed,
                       const float* __restrict__ conv_w,
                       float* __restrict__ qf,
                       float* __restrict__ kf,
                       float* __restrict__ vf)
{
    __shared__ float red[128];
    const int t   = blockIdx.x;
    const int grp = blockIdx.y;
    const int c   = threadIdx.x;

    int ch;
    if (grp < 16)       ch = grp * HEAD + c;                       // q
    else if (grp < 32)  ch = KEY_DIM + (grp - 16) * HEAD + c;      // k
    else                ch = 2 * KEY_DIM + (grp - 32) * HEAD + c;  // v

    // causal depthwise conv, kernel 4, left pad 3
    float y = 0.f;
#pragma unroll
    for (int i = 0; i < 4; ++i) {
        const int tt = t - 3 + i;
        if (tt >= 0) y += mixed[(size_t)tt * QKVZ_ROWS + ch] * conv_w[ch * 4 + i];
    }
    const float s = y / (1.f + __expf(-y));   // SiLU

    if (grp < 32) {
        // l2norm over the 128-wide head
        red[c] = s * s;
        __syncthreads();
        for (int off = 64; off > 0; off >>= 1) {
            if (c < off) red[c] += red[c + off];
            __syncthreads();
        }
        const float r = rsqrtf(red[0] + 1e-6f);
        if (grp < 16) {
            const float val = s * r * 0.08838834764831845f;    // * HEAD^-0.5
            const int vh = grp * 2;                            // repeat x2
            qf[((size_t)t * NUM_V_HEADS + vh)     * HEAD + c] = val;
            qf[((size_t)t * NUM_V_HEADS + vh + 1) * HEAD + c] = val;
        } else {
            const float val = s * r;
            const int vh = (grp - 16) * 2;
            kf[((size_t)t * NUM_V_HEADS + vh)     * HEAD + c] = val;
            kf[((size_t)t * NUM_V_HEADS + vh + 1) * HEAD + c] = val;
        }
    } else {
        vf[((size_t)t * NUM_V_HEADS + (grp - 32)) * HEAD + c] = s;
    }
}

// ---------------- gated delta-net recurrent scan ----------------------------
// One workgroup per v-head. State S[128][129] (pitch 129 -> conflict-free
// column reads, stride mod 64 == 1) lives in dynamic LDS.
__global__ void __launch_bounds__(256, 1) k_scan(const float* __restrict__ qf,
                                                 const float* __restrict__ kf,
                                                 const float* __restrict__ vf,
                                                 const float* __restrict__ g,
                                                 const float* __restrict__ beta,
                                                 float* __restrict__ obuf)
{
    extern __shared__ float smem[];
    float* S  = smem;                 // 128 * 129
    float* vK = S  + 128 * 129;
    float* vQ = vK + 128;
    float* vD = vQ + 128;

    const int h   = blockIdx.x;
    const int tid = threadIdx.x;

    for (int i = tid; i < 128 * 129; i += 256) S[i] = 0.f;
    __syncthreads();

    for (int t = 0; t < T_SEQ; ++t) {
        const size_t base  = ((size_t)t * NUM_V_HEADS + h) * HEAD;
        const float  decay = __expf(g[(size_t)t * NUM_V_HEADS + h]);
        const float  bt    = beta[(size_t)t * NUM_V_HEADS + h];
        if (tid < 128) { vK[tid] = kf[base + tid]; vQ[tid] = qf[base + tid]; }
        __syncthreads();

        // S *= exp(g_t)
        for (int i = tid; i < 128 * 128; i += 256) {
            const int kk = i >> 7, vv = i & 127;
            S[kk * 129 + vv] *= decay;
        }
        __syncthreads();

        // kv = S^T k ; delta = (v - kv) * beta
        if (tid < 128) {
            float a = 0.f;
#pragma unroll 8
            for (int kk = 0; kk < 128; ++kk) a += S[kk * 129 + tid] * vK[kk];
            vD[tid] = (vf[base + tid] - a) * bt;
        }
        __syncthreads();

        // S += outer(k, delta)
        for (int i = tid; i < 128 * 128; i += 256) {
            const int kk = i >> 7, vv = i & 127;
            S[kk * 129 + vv] += vK[kk] * vD[vv];
        }
        __syncthreads();

        // o = S^T q
        if (tid < 128) {
            float a = 0.f;
#pragma unroll 8
            for (int kk = 0; kk < 128; ++kk) a += S[kk * 129 + tid] * vQ[kk];
            obuf[base + tid] = a;
        }
        __syncthreads();
    }
}

// ---------------- RMS norm * norm_weight * SiLU(z), cast -> bf16 ------------
// grid = (T, 32), block = 128
__global__ void k_gate_norm(const float* __restrict__ obuf,
                            const float* __restrict__ mixed,
                            const float* __restrict__ norm_w,
                            __bf16* __restrict__ o_bf)
{
    __shared__ float red[128];
    const int t  = blockIdx.x;
    const int vh = blockIdx.y;
    const int c  = threadIdx.x;

    const float o = obuf[((size_t)t * NUM_V_HEADS + vh) * HEAD + c];
    red[c] = o * o;
    __syncthreads();
    for (int off = 64; off > 0; off >>= 1) {
        if (c < off) red[c] += red[c + off];
        __syncthreads();
    }
    const float rms = rsqrtf(red[0] * (1.f / HEAD) + EPS);
    const float z   = mixed[(size_t)t * QKVZ_ROWS + CONV_DIM + vh * HEAD + c];
    const float sz  = z / (1.f + __expf(-z));        // SiLU(z)
    const float res = o * rms * norm_w[c] * sz;
    o_bf[(size_t)t * VALUE_DIM + vh * HEAD + c] = (__bf16)res;
}

// ---------------------------------------------------------------------------
extern "C" void kernel_launch(void* const* d_in, const int* in_sizes, int n_in,
                              void* d_out, int out_size, void* d_ws, size_t ws_size,
                              hipStream_t stream)
{
    const float* hs      = (const float*)d_in[0];
    const float* W_qkvz  = (const float*)d_in[1];
    const float* W_ba    = (const float*)d_in[2];
    const float* conv_w  = (const float*)d_in[3];
    const float* dt_bias = (const float*)d_in[4];
    const float* A_log   = (const float*)d_in[5];
    const float* norm_w  = (const float*)d_in[6];
    const float* W_out   = (const float*)d_in[7];
    float* out = (float*)d_out;

    // workspace carve-out (all sizes are multiples of 256B already)
    char* p = (char*)d_ws;
    auto carve = [&](size_t bytes) { void* r = (void*)p; p += (bytes + 255) & ~(size_t)255; return r; };

    __bf16* hs_bf    = (__bf16*)carve((size_t)T_SEQ * HIDDEN * 2);
    __bf16* wqkvz_bf = (__bf16*)carve((size_t)QKVZ_ROWS * HIDDEN * 2);
    __bf16* wout_bf  = (__bf16*)carve((size_t)HIDDEN * VALUE_DIM * 2);
    float*  mixed    = (float*) carve((size_t)T_SEQ * QKVZ_ROWS * 4);
    float*  qf       = (float*) carve((size_t)T_SEQ * VALUE_DIM * 4);
    float*  kf       = (float*) carve((size_t)T_SEQ * VALUE_DIM * 4);
    float*  vf       = (float*) carve((size_t)T_SEQ * VALUE_DIM * 4);
    float*  gbuf     = (float*) carve((size_t)T_SEQ * NUM_V_HEADS * 4);
    float*  bbuf     = (float*) carve((size_t)T_SEQ * NUM_V_HEADS * 4);
    float*  obuf     = (float*) carve((size_t)T_SEQ * VALUE_DIM * 4);
    __bf16* o_bf     = (__bf16*)carve((size_t)T_SEQ * VALUE_DIM * 2);

    // 1) bf16 casts for the two big GEMMs
    k_cast_bf16<<<2048, 256, 0, stream>>>(hs,     hs_bf,    (size_t)T_SEQ * HIDDEN);
    k_cast_bf16<<<2048, 256, 0, stream>>>(W_qkvz, wqkvz_bf, (size_t)QKVZ_ROWS * HIDDEN);
    k_cast_bf16<<<2048, 256, 0, stream>>>(W_out,  wout_bf,  (size_t)HIDDEN * VALUE_DIM);

    // 2) mixed = hs @ W_qkvz^T   (WMMA bf16, f32 accumulate, async B staging)
    k_gemm_bf16<<<dim3(QKVZ_ROWS / 64, T_SEQ / 128), 256, 0, stream>>>(
        hs_bf, wqkvz_bf, mixed, T_SEQ, QKVZ_ROWS, HIDDEN);

    // 3) beta / g gates from ba = hs @ W_ba^T
    k_ba_gates<<<T_SEQ, 2 * NUM_V_HEADS, 0, stream>>>(hs, W_ba, dt_bias, A_log, bbuf, gbuf);

    // 4) conv + SiLU + l2norm + repeat -> q,k,v
    k_prep<<<dim3(T_SEQ, 64), HEAD, 0, stream>>>(mixed, conv_w, qf, kf, vf);

    // 5) recurrent gated delta-net scan (state in LDS)
    const size_t scan_lds = (size_t)(128 * 129 + 3 * 128) * sizeof(float);
    k_scan<<<NUM_V_HEADS, 256, scan_lds, stream>>>(qf, kf, vf, gbuf, bbuf, obuf);

    // 6) RMS norm + gate, cast to bf16
    k_gate_norm<<<dim3(T_SEQ, NUM_V_HEADS), HEAD, 0, stream>>>(obuf, mixed, norm_w, o_bf);

    // 7) out = o @ W_out^T   (WMMA bf16, f32 accumulate, f32 output)
    k_gemm_bf16<<<dim3(HIDDEN / 64, T_SEQ / 128), 256, 0, stream>>>(
        o_bf, wout_bf, out, T_SEQ, HIDDEN, VALUE_DIM);
}
